// TransitionUpdate_64673617543301
// MI455X (gfx1250) — 8-bit matrix operand order repaired and checked by reading, compiled, not run on hardware
//
#include <hip/hip_runtime.h>

typedef int   v16i __attribute__((ext_vector_type(16)));
typedef int   v4i  __attribute__((ext_vector_type(4)));
typedef float v8f  __attribute__((ext_vector_type(8)));
typedef float v4f  __attribute__((ext_vector_type(4)));

#define FP8_MAX 448.0f

__device__ __forceinline__ unsigned int sw_f32_to_e4m3(float f) {
  unsigned int u = __float_as_uint(f);
  unsigned int sign = (u >> 31) << 7;
  float a = fabsf(f);
  if (!(a > 0.f)) return sign;
  if (a >= 464.f) return sign | 0x7Eu;
  int e; (void)frexpf(a, &e);
  int E = (e - 1) + 7;
  if (E <= 0) {
    int mant = __float2int_rn(a * 512.f);
    if (mant >= 8) return sign | 0x08u;
    return sign | (unsigned int)mant;
  }
  int mant = __float2int_rn(ldexpf(a, 3 - (e - 1)));
  if (mant == 16) { mant = 8; ++E; if (E >= 16) return sign | 0x7Eu; }
  return sign | ((unsigned int)E << 3) | ((unsigned int)mant & 7u);
}

#if __has_builtin(__builtin_amdgcn_cvt_pk_fp8_f32)
__device__ __forceinline__ unsigned int pack4_fp8(float f0, float f1, float f2, float f3) {
  int d = __builtin_amdgcn_cvt_pk_fp8_f32(f0, f1, 0, false);
  d = __builtin_amdgcn_cvt_pk_fp8_f32(f2, f3, d, true);
  return (unsigned int)d;
}
#else
__device__ __forceinline__ unsigned int pack4_fp8(float f0, float f1, float f2, float f3) {
  return sw_f32_to_e4m3(f0) | (sw_f32_to_e4m3(f1) << 8) |
         (sw_f32_to_e4m3(f2) << 16) | (sw_f32_to_e4m3(f3) << 24);
}
#endif

__device__ __forceinline__ float bf16_lo(unsigned int w) { return __uint_as_float(w << 16); }
__device__ __forceinline__ float bf16_hi(unsigned int w) { return __uint_as_float(w & 0xFFFF0000u); }

__device__ __forceinline__ float wave_sum32(float v) {
#pragma unroll
  for (int m = 16; m >= 1; m >>= 1) v += __shfl_xor(v, m, 32);
  return v;
}
__device__ __forceinline__ float wave_max32(float v) {
#pragma unroll
  for (int m = 16; m >= 1; m >>= 1) v = fmaxf(v, __shfl_xor(v, m, 32));
  return v;
}

__device__ __forceinline__ float sel_row_scale(const float scv[8], const float sco[8],
                                               int nn, int hh) {
  const bool own = ((nn >> 3) == hh);
  float s0 = own ? scv[0] : sco[0], s1 = own ? scv[1] : sco[1];
  float s2 = own ? scv[2] : sco[2], s3 = own ? scv[3] : sco[3];
  float s4 = own ? scv[4] : sco[4], s5 = own ? scv[5] : sco[5];
  float s6 = own ? scv[6] : sco[6], s7 = own ? scv[7] : sco[7];
  float t0 = (nn & 1) ? s1 : s0, t1 = (nn & 1) ? s3 : s2;
  float t2 = (nn & 1) ? s5 : s4, t3 = (nn & 1) ? s7 : s6;
  float u0 = (nn & 2) ? t1 : t0, u1 = (nn & 2) ? t3 : t2;
  return (nn & 4) ? u1 : u0;
}

__global__ void fp8_quant_rows_kernel(const float* __restrict__ src,
                                      unsigned char* __restrict__ q,
                                      float* __restrict__ scales,
                                      int K, int nrows) {
  const int wid  = threadIdx.x >> 5;
  const int lane = threadIdx.x & 31;
  const int row  = blockIdx.x * 8 + wid;
  if (row >= nrows) return;
  const float* r = src + (long)row * K;
  const int chunks = K >> 7;
  v4f vals[4];
  float amax = 0.f;
#pragma unroll
  for (int i = 0; i < 4; ++i) {
    if (i >= chunks) break;
    v4f v = *(const v4f*)(r + i * 128 + lane * 4);
    vals[i] = v;
    amax = fmaxf(amax, fmaxf(fmaxf(fabsf(v.x), fabsf(v.y)), fmaxf(fabsf(v.z), fabsf(v.w))));
  }
  amax = wave_max32(amax);
  const float scale = fmaxf(amax, 1e-12f) * (1.f / FP8_MAX);
  const float inv = 1.f / scale;
  unsigned int* qd = (unsigned int*)(q + (long)row * K);
#pragma unroll
  for (int i = 0; i < 4; ++i) {
    if (i >= chunks) break;
    v4f v = vals[i];
    qd[i * 32 + lane] = pack4_fp8(v.x * inv, v.y * inv, v.z * inv, v.w * inv);
  }
  if (lane == 0) scales[row] = scale;
}

__global__ __launch_bounds__(128) void fused_ln_fp8mlp_kernel(
    const float* __restrict__ x,
    const float* __restrict__ lnw,
    const float* __restrict__ lnb,
    const unsigned char* __restrict__ wq,
    const float* __restrict__ sw1,
    const float* __restrict__ sw2,
    float* __restrict__ out) {
  extern __shared__ char smem[];
  const int tid  = threadIdx.x;
  const int wid  = tid >> 5;
  const int lane = tid & 31;
  const int hh   = lane >> 4;
  const int nn   = lane & 15;
  const long row_base = (long)blockIdx.x * 128 + wid * 32;

  char*  lds_w1 = smem;
  char*  lds_w2 = smem + 131072;
  char*  gb     = smem + 196608 + wid * 32768;
  char*  hq     = gb;
  float* sxl    = (float*)(gb + 4096);

  {
    const v4i* src = (const v4i*)wq;
    v4i* dst = (v4i*)smem;
#pragma unroll 4
    for (int i = 0; i < 96; ++i)
      dst[tid + i * 128] = src[tid + i * 128];
  }

  const v4f lw = *(const v4f*)(lnw + lane * 4);
  const v4f lb = *(const v4f*)(lnb + lane * 4);
  for (int r = 0; r < 32; ++r) {
    v4f xv = *(const v4f*)(x + (row_base + r) * 128 + lane * 4);
    float mu = wave_sum32(xv.x + xv.y + xv.z + xv.w) * (1.f / 128.f);
    v4f d; d.x = xv.x - mu; d.y = xv.y - mu; d.z = xv.z - mu; d.w = xv.w - mu;
    float var = wave_sum32(d.x * d.x + d.y * d.y + d.z * d.z + d.w * d.w) * (1.f / 128.f);
    float rs = rsqrtf(var + 1e-5f);
    v4f h;
    h.x = d.x * rs * lw.x + lb.x;  h.y = d.y * rs * lw.y + lb.y;
    h.z = d.z * rs * lw.z + lb.z;  h.w = d.w * rs * lw.w + lb.w;
    float amax = fmaxf(fmaxf(fabsf(h.x), fabsf(h.y)), fmaxf(fabsf(h.z), fabsf(h.w)));
    amax = wave_max32(amax);
    float scale = fmaxf(amax, 1e-12f) * (1.f / FP8_MAX);
    float inv = 1.f / scale;
    *(unsigned int*)(hq + r * 128 + lane * 4) = pack4_fp8(h.x * inv, h.y * inv, h.z * inv, h.w * inv);
    if (lane == 0) sxl[r] = scale;
  }

  float sxr0[8], sxr1[8];
#pragma unroll
  for (int v = 0; v < 8; ++v) { sxr0[v] = sxl[v + 8 * hh]; sxr1[v] = sxl[16 + v + 8 * hh]; }

  v16i A1_0, A1_1;
#pragma unroll
  for (int b = 0; b < 2; ++b)
#pragma unroll
    for (int j = 0; j < 4; ++j) {
      const int off = b * 64 + (j >> 1) * 32 + hh * 16 + (j & 1) * 8;
      unsigned long long q0 = *(const unsigned long long*)(hq + nn * 128 + off);
      unsigned long long q1 = *(const unsigned long long*)(hq + (16 + nn) * 128 + off);
      const int d = b * 8 + j * 2;
      A1_0[d] = (int)(unsigned int)q0;  A1_0[d + 1] = (int)(q0 >> 32);
      A1_1[d] = (int)(unsigned int)q1;  A1_1[d + 1] = (int)(q1 >> 32);
    }

  __syncthreads();

  float amax0[8], amax1[8];
#pragma unroll
  for (int v = 0; v < 8; ++v) { amax0[v] = 0.f; amax1[v] = 0.f; }

  for (int t = 0; t < 32; ++t) {
    v16i Ba, Bb;
    const char* pa = lds_w1 + (t * 16 + nn) * 128 + hh * 16;
    const char* pb = lds_w1 + ((512 + t * 16) + nn) * 128 + hh * 16;
#pragma unroll
    for (int g = 0; g < 4; ++g) {
      v4i da = *(const v4i*)(pa + g * 32);
      v4i db = *(const v4i*)(pb + g * 32);
      Ba[4 * g] = da.x; Ba[4 * g + 1] = da.y; Ba[4 * g + 2] = da.z; Ba[4 * g + 3] = da.w;
      Bb[4 * g] = db.x; Bb[4 * g + 1] = db.y; Bb[4 * g + 2] = db.z; Bb[4 * g + 3] = db.w;
    }
    v8f ca0 = {}, cb0 = {}, ca1 = {}, cb1 = {};
    ca0 = __builtin_amdgcn_wmma_f32_16x16x128_fp8_fp8(A1_0, Ba, (short)0, ca0, false, false);
    cb0 = __builtin_amdgcn_wmma_f32_16x16x128_fp8_fp8(A1_0, Bb, (short)0, cb0, false, false);
    ca1 = __builtin_amdgcn_wmma_f32_16x16x128_fp8_fp8(A1_1, Ba, (short)0, ca1, false, false);
    cb1 = __builtin_amdgcn_wmma_f32_16x16x128_fp8_fp8(A1_1, Bb, (short)0, cb1, false, false);
    const float swa = sw1[t * 16 + nn];
    const float swb = sw1[512 + t * 16 + nn];
#pragma unroll
    for (int v = 0; v < 8; ++v) {
      float g0 = fmaxf(ca0[v] * sxr0[v] * swa, 0.f) * (cb0[v] * sxr0[v] * swb);
      float g1 = fmaxf(ca1[v] * sxr1[v] * swa, 0.f) * (cb1[v] * sxr1[v] * swb);
      amax0[v] = fmaxf(amax0[v], fabsf(g0));
      amax1[v] = fmaxf(amax1[v], fabsf(g1));
      *(unsigned short*)(gb + (v + 8 * hh) * 1024 + (t * 16 + nn) * 2) =
          (unsigned short)(__float_as_uint(g0) >> 16);
      *(unsigned short*)(gb + (16 + v + 8 * hh) * 1024 + (t * 16 + nn) * 2) =
          (unsigned short)(__float_as_uint(g1) >> 16);
    }
  }

#pragma unroll
  for (int v = 0; v < 8; ++v)
#pragma unroll
    for (int m = 8; m >= 1; m >>= 1) {
      amax0[v] = fmaxf(amax0[v], __shfl_xor(amax0[v], m, 32));
      amax1[v] = fmaxf(amax1[v], __shfl_xor(amax1[v], m, 32));
    }
  float scv0[8], sco0[8], scv1[8], sco1[8];
#pragma unroll
  for (int v = 0; v < 8; ++v) {
    scv0[v] = fmaxf(amax0[v], 1e-12f) * (1.f / FP8_MAX);
    scv1[v] = fmaxf(amax1[v], 1e-12f) * (1.f / FP8_MAX);
  }
#pragma unroll
  for (int v = 0; v < 8; ++v) {
    sco0[v] = __shfl_xor(scv0[v], 16, 32);
    sco1[v] = __shfl_xor(scv1[v], 16, 32);
  }
  const float invsg0 = 1.f / sel_row_scale(scv0, sco0, nn, hh);
  const float invsg1 = 1.f / sel_row_scale(scv1, sco1, nn, hh);

  v16i A2_0[4], A2_1[4];
#pragma unroll
  for (int kc = 0; kc < 4; ++kc)
#pragma unroll
    for (int b = 0; b < 2; ++b)
#pragma unroll
      for (int j = 0; j < 4; ++j) {
        const int K0 = kc * 128 + b * 64 + (j >> 1) * 32 + hh * 16 + (j & 1) * 8;
        const int d = b * 8 + j * 2;
        v4i w0 = *(const v4i*)(gb + nn * 1024 + K0 * 2);
        v4i w1v = *(const v4i*)(gb + (16 + nn) * 1024 + K0 * 2);
        A2_0[kc][d] = (int)pack4_fp8(bf16_lo((unsigned)w0.x) * invsg0, bf16_hi((unsigned)w0.x) * invsg0,
                                     bf16_lo((unsigned)w0.y) * invsg0, bf16_hi((unsigned)w0.y) * invsg0);
        A2_0[kc][d + 1] = (int)pack4_fp8(bf16_lo((unsigned)w0.z) * invsg0, bf16_hi((unsigned)w0.z) * invsg0,
                                         bf16_lo((unsigned)w0.w) * invsg0, bf16_hi((unsigned)w0.w) * invsg0);
        A2_1[kc][d] = (int)pack4_fp8(bf16_lo((unsigned)w1v.x) * invsg1, bf16_hi((unsigned)w1v.x) * invsg1,
                                     bf16_lo((unsigned)w1v.y) * invsg1, bf16_hi((unsigned)w1v.y) * invsg1);
        A2_1[kc][d + 1] = (int)pack4_fp8(bf16_lo((unsigned)w1v.z) * invsg1, bf16_hi((unsigned)w1v.z) * invsg1,
                                         bf16_lo((unsigned)w1v.w) * invsg1, bf16_hi((unsigned)w1v.w) * invsg1);
      }

  for (int t = 0; t < 8; ++t) {
    v8f c0 = {}, c1 = {};
#pragma unroll
    for (int kc = 0; kc < 4; ++kc) {
      v16i B;
      const char* p = lds_w2 + (t * 16 + nn) * 512 + kc * 128 + hh * 16;
#pragma unroll
      for (int g = 0; g < 4; ++g) {
        v4i dv = *(const v4i*)(p + g * 32);
        B[4 * g] = dv.x; B[4 * g + 1] = dv.y; B[4 * g + 2] = dv.z; B[4 * g + 3] = dv.w;
      }
      c0 = __builtin_amdgcn_wmma_f32_16x16x128_fp8_fp8(A2_0[kc], B, (short)0, c0, false, false);
      c1 = __builtin_amdgcn_wmma_f32_16x16x128_fp8_fp8(A2_1[kc], B, (short)0, c1, false, false);
    }
    const float s2o = sw2[t * 16 + nn];
#pragma unroll
    for (int v = 0; v < 8; ++v) {
      out[(row_base + v + 8 * hh) * 128 + t * 16 + nn]      = c0[v] * scv0[v] * s2o;
      out[(row_base + 16 + v + 8 * hh) * 128 + t * 16 + nn] = c1[v] * scv1[v] * s2o;
    }
  }
}

extern "C" void kernel_launch(void* const* d_in, const int* in_sizes, int n_in,
                              void* d_out, int out_size, void* d_ws, size_t ws_size,
                              hipStream_t stream) {
  (void)n_in; (void)out_size; (void)ws_size;
  const float* x   = (const float*)d_in[0];
  const float* lnw = (const float*)d_in[1];
  const float* lnb = (const float*)d_in[2];
  const float* w1  = (const float*)d_in[3];
  const float* w2  = (const float*)d_in[4];

  unsigned char* w1q = (unsigned char*)d_ws;
  unsigned char* w2q = w1q + 1024 * 128;
  float* sw1 = (float*)(w2q + 128 * 512);
  float* sw2 = sw1 + 1024;

  fp8_quant_rows_kernel<<<1024 / 8, 256, 0, stream>>>(w1, w1q, sw1, 128, 1024);
  fp8_quant_rows_kernel<<<128 / 8, 256, 0, stream>>>(w2, w2q, sw2, 512, 128);

  const int rows = in_sizes[0] / 128;
  const size_t smem = 327680;
  (void)hipFuncSetAttribute((const void*)fused_ln_fp8mlp_kernel,
                            hipFuncAttributeMaxDynamicSharedMemorySize, (int)smem);
  fused_ln_fp8mlp_kernel<<<rows / 128, 128, smem, stream>>>(
      x, lnw, lnb, w1q, sw1, sw2, (float*)d_out);
}
